// EMAVectorQuantizer_26551487824053
// MI455X (gfx1250) — matrix operand order repaired and checked by reading, not compiled, not run on hardware
//
#include <hip/hip_runtime.h>
#include <hip/hip_bf16.h>

// ---------------------------------------------------------------------------
// EMA Vector Quantizer for MI455X (gfx1250), wave32 + WMMA bf16 + TDM staging.
// x[32768,64], embedding[1024,64], cluster_size[1024], ema_w[1024,64].
// Outputs (flat, in order): quantized(2097152), vq_loss(1), indices(32768),
// new_cluster_size(1024), new_ema_w(65536), new_embedding(65536)
// ---------------------------------------------------------------------------

#define NTOK   32768
#define DDIM   64
#define KCODE  1024
#define ROWS_PER_WG 128           // 8 waves * 16 rows
#define EPAD_HALFS 72             // 64 bf16 + 8 pad -> 144B stride (conflict-free b128)

typedef __bf16 v16bf __attribute__((ext_vector_type(16)));
typedef float  v8f   __attribute__((ext_vector_type(8)));
typedef unsigned int uint32x4 __attribute__((ext_vector_type(4)));
typedef int          int32x4  __attribute__((ext_vector_type(4)));
typedef int          int32x8  __attribute__((ext_vector_type(8)));

union BF16x16 { v16bf v; unsigned short s[16]; };

#if defined(__gfx1250__) && __has_builtin(__builtin_amdgcn_tensor_load_to_lds)
#define USE_TDM 1
#warning "CDNA5 probe: tensor_load_to_lds builtin AVAILABLE - TDM staging path enabled"
#else
#if defined(__gfx1250__)
#warning "CDNA5 probe: tensor_load_to_lds builtin NOT available - fallback LDS staging"
#endif
#endif

__device__ __forceinline__ unsigned short f32_bf16(float f) {
    unsigned u = __float_as_uint(f);
    u += 0x7FFFu + ((u >> 16) & 1u);          // round-to-nearest-even
    return (unsigned short)(u >> 16);
}

// convert float4 -> 4 bf16 with scale (-2 for the A matrix)
__device__ __forceinline__ void cvt4s(const float4 f, float s, unsigned short* d) {
    d[0] = f32_bf16(s * f.x); d[1] = f32_bf16(s * f.y);
    d[2] = f32_bf16(s * f.z); d[3] = f32_bf16(s * f.w);
}

// ----------------- kernel 1: ||e_k||^2 + bf16 codebook to ws ---------------
__global__ __launch_bounds__(64)
void vq_prep(const float* __restrict__ emb, unsigned short* __restrict__ ebf16,
             float* __restrict__ enorm, float* __restrict__ loss_acc) {
    __shared__ float red[64];
    const int k = blockIdx.x, t = threadIdx.x;
    const float v = emb[k * DDIM + t];
    ebf16[k * DDIM + t] = f32_bf16(v);
    red[t] = v * v;
    __syncthreads();
    for (int s = 32; s > 0; s >>= 1) {
        if (t < s) red[t] += red[t + s];
        __syncthreads();
    }
    if (t == 0) enorm[k] = red[0];
    if (k == 0 && t == 0) loss_acc[0] = 0.0f;
}

// --------------- kernel 2: seed EMA outputs with decay term ----------------
__global__ __launch_bounds__(256)
void vq_init_ema(const float* __restrict__ cs, const float* __restrict__ ew,
                 float* __restrict__ out_cs, float* __restrict__ out_ew) {
    const int i = blockIdx.x * 256 + threadIdx.x;
    if (i < KCODE)        out_cs[i] = 0.99f * cs[i];
    if (i < KCODE * DDIM) out_ew[i] = 0.99f * ew[i];
}

// ------------- kernel 3: WMMA distances + argmin + gather + loss -----------
extern __shared__ __align__(16) unsigned char smem[];

__global__ __launch_bounds__(256)
void vq_main(const float* __restrict__ x, const float* __restrict__ emb,
             const unsigned short* __restrict__ ebf16_ws,
             const float* __restrict__ enorm_g,
             float* __restrict__ out_q, float* __restrict__ out_idx,
             float* __restrict__ loss_acc) {
    unsigned short* eb   = (unsigned short*)smem;                    // 1024*144 B
    float*          enrm = (float*)(smem + KCODE * EPAD_HALFS * 2);  // 4 KB
    int*            idxb = (int*)(smem + KCODE * EPAD_HALFS * 2 + 4096);   // 512 B
    float*          red  = (float*)(smem + KCODE * EPAD_HALFS * 2 + 4608); // 1 KB

    const int tid = threadIdx.x;

#ifdef USE_TDM
    // ---- one Tensor-DMA moves the whole bf16 codebook into LDS, padded ----
    if (tid < 32) {                          // wave 0 issues the DMA
        const unsigned lds_off = (unsigned)(unsigned long long)(const void*)eb;
        const unsigned long long ga = (unsigned long long)ebf16_ws;
        uint32x4 g0;
        g0[0] = 1u;                                   // count=1 (valid, user mode)
        g0[1] = lds_off;                              // lds_addr
        g0[2] = (unsigned)(ga & 0xFFFFFFFFull);       // global_addr[31:0]
        g0[3] = (unsigned)((ga >> 32) & 0x1FFFFFFull) | (2u << 30);  // [56:32] | type=2
        int32x8 g1;
        g1[0] = (1 << 16)            // data_size = 2 bytes
              | (1 << 20)            // pad_enable
              | (4 << 22)            // pad_interval: 32 DWORDs (=128B row)
              | (3 << 25);           // pad_amount: 4 DWORDs (=16B pad)
        g1[1] = (DDIM << 16);        // tensor_dim0[15:0] = 64
        g1[2] = (KCODE << 16);       // tensor_dim1[15:0] = 1024
        g1[3] = (DDIM << 16);        // tile_dim0 = 64
        g1[4] = KCODE;               // tile_dim1 = 1024, tile_dim2 = 0
        g1[5] = DDIM;                // tensor_dim0_stride = 64 elements
        g1[6] = 0;
        g1[7] = 0;
        int32x4 z4; z4[0] = 0; z4[1] = 0; z4[2] = 0; z4[3] = 0;
#if __clang_major__ >= 23
        int32x8 z8; z8[0]=0; z8[1]=0; z8[2]=0; z8[3]=0; z8[4]=0; z8[5]=0; z8[6]=0; z8[7]=0;
        __builtin_amdgcn_tensor_load_to_lds(g0, g1, z4, z4, z8, 0);
#else
        __builtin_amdgcn_tensor_load_to_lds(g0, g1, z4, z4, 0);
#endif
        __builtin_amdgcn_s_wait_tensorcnt(0);
    }
#else
    // ---- fallback: cooperative bf16 copy from ws into padded LDS rows ----
    for (int i = tid; i < KCODE * 8; i += 256) {      // 16B granules, 8 per row
        const int row = i >> 3, c8 = i & 7;
        const uint4 v = ((const uint4*)ebf16_ws)[i];
        *(uint4*)(eb + row * EPAD_HALFS + c8 * 8) = v;
    }
#endif
    for (int i = tid; i < KCODE; i += 256) enrm[i] = enorm_g[i];
    __syncthreads();

    // ---- per-wave A tiles: 16 rows of x, scaled by -2, bf16 lane layout ----
    const int wave = tid >> 5, lane = tid & 31;
    const int m = lane & 15, h = lane >> 4;
    const long rowg = (long)blockIdx.x * ROWS_PER_WG + wave * 16 + m;
    const float* xr = x + rowg * DDIM;

    BF16x16 A0, A1;
    {
        const float4 f0 = *(const float4*)(xr + h * 8);
        const float4 f1 = *(const float4*)(xr + h * 8 + 4);
        const float4 f2 = *(const float4*)(xr + 16 + h * 8);
        const float4 f3 = *(const float4*)(xr + 16 + h * 8 + 4);
        const float4 f4 = *(const float4*)(xr + 32 + h * 8);
        const float4 f5 = *(const float4*)(xr + 32 + h * 8 + 4);
        const float4 f6 = *(const float4*)(xr + 48 + h * 8);
        const float4 f7 = *(const float4*)(xr + 48 + h * 8 + 4);
        cvt4s(f0, -2.0f, &A0.s[0]);  cvt4s(f1, -2.0f, &A0.s[4]);
        cvt4s(f2, -2.0f, &A0.s[8]);  cvt4s(f3, -2.0f, &A0.s[12]);
        cvt4s(f4, -2.0f, &A1.s[0]);  cvt4s(f5, -2.0f, &A1.s[4]);
        cvt4s(f6, -2.0f, &A1.s[8]);  cvt4s(f7, -2.0f, &A1.s[12]);
    }

    float best[8];
    int   bidx[8];
#pragma unroll
    for (int r = 0; r < 8; ++r) { best[r] = 3.4e38f; bidx[r] = 0; }
    const int n = lane & 15;

    // B-tile fetch (two K-chunks of column tile cc; cc wraps to stay in LDS)
    auto loadB = [&](BF16x16& L0, BF16x16& L1, int cc) {
        const int col = (cc & 63) * 16 + n;
        const unsigned short* bp = eb + col * EPAD_HALFS + h * 8;
        *(uint4*)&L0.s[0] = *(const uint4*)(bp);
        *(uint4*)&L0.s[8] = *(const uint4*)(bp + 16);
        *(uint4*)&L1.s[0] = *(const uint4*)(bp + 32);
        *(uint4*)&L1.s[8] = *(const uint4*)(bp + 48);
    };
    auto loadEn = [&](int cc) -> float { return enrm[(cc & 63) * 16 + n]; };

    // 4 WMMAs + argmin update for two column tiles
    auto process = [&](const BF16x16& P00, const BF16x16& P01,
                       const BF16x16& P10, const BF16x16& P11,
                       float en0, float en1, int c0) {
        const int col0 = c0 * 16 + n;
        const int col1 = col0 + 16;
        // C init = ||e||^2 splat; A carries -2, so D = ||e||^2 - 2*x.e
        v8f acc0 = {en0, en0, en0, en0, en0, en0, en0, en0};
        v8f acc1 = {en1, en1, en1, en1, en1, en1, en1, en1};
        acc0 = __builtin_amdgcn_wmma_f32_16x16x32_bf16(false, A0.v, false, P00.v,
                                                       (short)0, acc0, false, false);
        acc0 = __builtin_amdgcn_wmma_f32_16x16x32_bf16(false, A1.v, false, P01.v,
                                                       (short)0, acc0, false, false);
        acc1 = __builtin_amdgcn_wmma_f32_16x16x32_bf16(false, A0.v, false, P10.v,
                                                       (short)0, acc1, false, false);
        acc1 = __builtin_amdgcn_wmma_f32_16x16x32_bf16(false, A1.v, false, P11.v,
                                                       (short)0, acc1, false, false);
#pragma unroll
        for (int r = 0; r < 8; ++r) {
            const float v0 = acc0[r];
            if (v0 < best[r]) { best[r] = v0; bidx[r] = col0; }
        }
#pragma unroll
        for (int r = 0; r < 8; ++r) {
            const float v1 = acc1[r];
            if (v1 < best[r]) { best[r] = v1; bidx[r] = col1; }
        }
    };

    // ---- software-pipelined sweep: prefetch distance = 2 column tiles ----
    BF16x16 Ba00, Ba01, Ba10, Ba11;      // buffer A: tiles c, c+1
    BF16x16 Bb00, Bb01, Bb10, Bb11;      // buffer B: tiles c+2, c+3
    float enA0 = loadEn(0), enA1 = loadEn(1);
    loadB(Ba00, Ba01, 0);
    loadB(Ba10, Ba11, 1);
    for (int c = 0; c < KCODE / 16; c += 4) {
        loadB(Bb00, Bb01, c + 2);
        loadB(Bb10, Bb11, c + 3);
        const float enB0 = loadEn(c + 2), enB1 = loadEn(c + 3);
        process(Ba00, Ba01, Ba10, Ba11, enA0, enA1, c);
        loadB(Ba00, Ba01, c + 4);
        loadB(Ba10, Ba11, c + 5);
        enA0 = loadEn(c + 4); enA1 = loadEn(c + 5);
        process(Bb00, Bb01, Bb10, Bb11, enB0, enB1, c + 2);
    }

    // ---- branch-free 16-lane min reduce; tie-break -> smaller index ----
#pragma unroll
    for (int r = 0; r < 8; ++r) {
        float bv = best[r];
        int   bi = bidx[r];
#pragma unroll
        for (int s = 1; s < 16; s <<= 1) {
            const float ov = __shfl_xor(bv, s, 16);
            const int   oi = __shfl_xor(bi, s, 16);
            const bool  tk = (ov < bv) | ((ov == bv) & (oi < bi));
            bi = tk ? oi : bi;
            bv = tk ? ov : bv;
        }
        if (n == 0) {
            const int mrow = wave * 16 + h * 8 + r;    // D layout: M = h*8 + r
            idxb[mrow] = bi;
            out_idx[(long)blockIdx.x * ROWS_PER_WG + mrow] = (float)bi;
        }
    }
    __syncthreads();

    // ---- gather quantized rows (fp32, exact) + loss partial ----
    float lsum = 0.0f;
    for (int j = tid; j < ROWS_PER_WG * (DDIM / 4); j += 256) {
        const int row = j >> 4, c4 = j & 15;
        const int k = idxb[row];
        const float4 q = *(const float4*)(emb + (long)k * DDIM + c4 * 4);
        const long g = ((long)blockIdx.x * ROWS_PER_WG + row) * DDIM + c4 * 4;
        const float4 xv = *(const float4*)(x + g);
        *(float4*)(out_q + g) = q;
        const float dx = q.x - xv.x, dy = q.y - xv.y;
        const float dz = q.z - xv.z, dw = q.w - xv.w;
        lsum += dx * dx + dy * dy + dz * dz + dw * dw;
    }
    red[tid] = lsum;
    __syncthreads();
    for (int s = 128; s > 0; s >>= 1) {
        if (tid < s) red[tid] += red[tid + s];
        __syncthreads();
    }
    if (tid == 0) atomicAdd(loss_acc, red[0]);
}

// --------------------- kernel 4: scatter counts + dw -----------------------
__global__ __launch_bounds__(256)
void vq_scatter(const float* __restrict__ x, const float* __restrict__ idxf,
                float* __restrict__ out_cs, float* __restrict__ out_ew) {
    const long t = (long)blockIdx.x * 256 + threadIdx.x;  // over N*D
    const int row = (int)(t >> 6);
    const int col = (int)(t & 63);
    const int k = (int)idxf[row];
    atomicAdd(&out_ew[k * DDIM + col], 0.01f * x[t]);
    if (col == 0) atomicAdd(&out_cs[k], 0.01f);
}

// ------------------- kernel 5: finalize embedding + loss -------------------
__global__ __launch_bounds__(1024)
void vq_final(const float* __restrict__ out_cs, const float* __restrict__ out_ew,
              float* __restrict__ out_emb, const float* __restrict__ loss_acc,
              float* __restrict__ out_loss) {
    __shared__ float red[1024];
    const int k = threadIdx.x;
    const float ncs = out_cs[k];
    red[k] = ncs;
    __syncthreads();
    for (int s = 512; s > 0; s >>= 1) {
        if (k < s) red[k] += red[k + s];
        __syncthreads();
    }
    const float ntot = red[0];
    const float cs = (ncs + 1e-5f) / (ntot + (float)KCODE * 1e-5f) * ntot;
    const float inv = 1.0f / cs;
    for (int d = 0; d < DDIM; ++d)
        out_emb[k * DDIM + d] = out_ew[k * DDIM + d] * inv;
    if (k == 0)
        out_loss[0] = 2.0f * loss_acc[0] / (float)((long)NTOK * DDIM);
}

// ---------------------------------------------------------------------------
extern "C" void kernel_launch(void* const* d_in, const int* in_sizes, int n_in,
                              void* d_out, int out_size, void* d_ws, size_t ws_size,
                              hipStream_t stream) {
    const float* x   = (const float*)d_in[0];   // [32768,64]
    const float* emb = (const float*)d_in[1];   // [1024,64]
    const float* cs  = (const float*)d_in[2];   // [1024]
    const float* ew  = (const float*)d_in[3];   // [1024,64]

    float* out      = (float*)d_out;
    float* out_q    = out;                                  // 2097152
    float* out_loss = out + (long)NTOK * DDIM;              // 1
    float* out_idx  = out_loss + 1;                         // 32768
    float* out_cs   = out_idx + NTOK;                       // 1024
    float* out_ew   = out_cs + KCODE;                       // 65536
    float* out_emb  = out_ew + KCODE * DDIM;                // 65536

    unsigned short* ws_ebf16 = (unsigned short*)d_ws;       // 65536 halfs (128 KB)
    float* ws_enorm = (float*)((char*)d_ws + KCODE * DDIM * 2);  // 1024 f
    float* ws_loss  = ws_enorm + KCODE;                     // 1 f

    const int SMEM = KCODE * EPAD_HALFS * 2 + 4096 + 512 + 1024;  // 153088 B
    static bool attr_set = false;
    if (!attr_set) {
        hipFuncSetAttribute((const void*)vq_main,
                            hipFuncAttributeMaxDynamicSharedMemorySize, SMEM);
        attr_set = true;
    }

    vq_prep<<<KCODE, 64, 0, stream>>>(emb, ws_ebf16, ws_enorm, ws_loss);
    vq_init_ema<<<(KCODE * DDIM + 255) / 256, 256, 0, stream>>>(cs, ew, out_cs, out_ew);
    vq_main<<<NTOK / ROWS_PER_WG, 256, SMEM, stream>>>(x, emb, ws_ebf16, ws_enorm,
                                                       out_q, out_idx, ws_loss);
    vq_scatter<<<((long)NTOK * DDIM) / 256, 256, 0, stream>>>(x, out_idx, out_cs, out_ew);
    vq_final<<<1, 1024, 0, stream>>>(out_cs, out_ew, out_emb, ws_loss, out_loss);
}
